// RpeSelfAttentionV2S1_30932354466495
// MI455X (gfx1250) — compile-verified, matrix operand order checked by reading, not run on hardware
//
#include <hip/hip_runtime.h>
#include <math.h>

// ---------------- problem constants (from reference) ----------------
#define E_DIM 1024
#define H_DIM 16
#define D_DIM 64
#define B_DIM 2
#define S_SUM 32
#define R_REG 2048
#define T_TOT (S_SUM + R_REG)   // 2080
#define SCALING 0.125f          // 64^-0.5

// ---------------- WMMA types / helpers ----------------
typedef __attribute__((ext_vector_type(16))) __bf16 v16bf;
typedef __attribute__((ext_vector_type(4)))  __bf16 v4bf;
typedef __attribute__((ext_vector_type(2)))  __bf16 v2bf;
typedef __attribute__((ext_vector_type(8)))  float  v8f;
typedef __attribute__((ext_vector_type(4)))  float  v4f;
typedef __attribute__((ext_vector_type(2)))  float  v2f;

struct U128x2 { uint4 lo; uint4 hi; };

__device__ __forceinline__ unsigned short f2bf(float f) {
  return __builtin_bit_cast(unsigned short, (__bf16)f);
}

// packed f32x2 -> bf16x2 (single v_cvt_pk_bf16_f32, no re-pack)
__device__ __forceinline__ unsigned int pack2(float a, float b) {
  v2f f = {a, b};
  v2bf h = __builtin_convertvector(f, v2bf);
  return __builtin_bit_cast(unsigned int, h);
}

// packed f32x4 -> bf16x4 (two v_cvt_pk_bf16_f32, stored as one b64)
__device__ __forceinline__ uint2 pack4(float4 v) {
  v4f f = {v.x, v.y, v.z, v.w};
  v4bf h = __builtin_convertvector(f, v4bf);
  return __builtin_bit_cast(uint2, h);
}

__device__ __forceinline__ v8f vzero8() {
  v8f z = {0.f, 0.f, 0.f, 0.f, 0.f, 0.f, 0.f, 0.f};
  return z;
}

// Load a 16x32 bf16 A/B fragment from LDS laid out row-major [row][k],
// following the CDNA5 16-bit operand striping:
//   lanes 0-15  : row = lane,    elems 0-7 -> K 0..7,  elems 8-15 -> K 16..23
//   lanes 16-31 : row = lane-16, elems 0-7 -> K 8..15, elems 8-15 -> K 24..31
__device__ __forceinline__ v16bf lds_frag(const unsigned short* base,
                                          int row0, int col0, int stride) {
  int lane = threadIdx.x & 31;
  int r = row0 + (lane & 15);
  int c = col0 + ((lane >> 4) << 3);
  const unsigned short* p = base + r * stride + c;
  U128x2 t;
  t.lo = *(const uint4*)(p);
  t.hi = *(const uint4*)(p + 16);
  return __builtin_bit_cast(v16bf, t);
}

__device__ __forceinline__ v8f wmma_bf16(v16bf a, v16bf b, v8f c) {
  // (neg_a, A, neg_b, B, c_mod, C, reuse_a, reuse_b)
  return __builtin_amdgcn_wmma_f32_16x16x32_bf16(false, a, false, b,
                                                 (short)0, c, false, false);
}

// -------- async global->LDS helpers (ASYNCcnt-tracked, gfx1250) --------
__device__ __forceinline__ unsigned lds_off32(const void* p) {
  // low 32 bits of an LDS generic address are the LDS byte offset
  return (unsigned)(unsigned long long)p;
}

__device__ __forceinline__ void async_load_b128(unsigned lds_byte_off,
                                                const float* gaddr) {
  asm volatile("global_load_async_to_lds_b128 %0, %1, off"
               :: "v"(lds_byte_off), "v"(gaddr)
               : "memory");
}

__device__ __forceinline__ void wait_async0() {
  asm volatile("s_wait_asynccnt 0x0" ::: "memory");
}

// ---------------- GEMM: Y[M,N] = A[M,K] * W[N,K]^T + bias[N] ----------------
// block tile 128x128x32, 256 threads (8 wave32s), bf16 WMMA, f32 accumulate,
// double-buffered LDS: issue loads -> compute current -> convert/store next.
#define GBM 128
#define GBN 128
#define GBK 32

__device__ __forceinline__ void gemm_issue_loads(
    const float* __restrict__ A, const float* __restrict__ W,
    int M, int K, int m0, int n0, int k0, int tid,
    float4 (&ra)[4], float4 (&rb)[4]) {
#pragma unroll
  for (int i = 0; i < 4; ++i) {            // 1024 float4 slots / 256 threads
    int idx = tid + (i << 8);
    int r = idx >> 3;                      // 8 float4 per 32-wide row
    int c4 = idx & 7;
    float4 z = {0.f, 0.f, 0.f, 0.f};
    ra[i] = (m0 + r < M)
                ? *(const float4*)(A + (size_t)(m0 + r) * K + k0 + (c4 << 2))
                : z;
    rb[i] = *(const float4*)(W + (size_t)(n0 + r) * K + k0 + (c4 << 2));
  }
}

__device__ __forceinline__ void gemm_store_tiles(
    unsigned short (*As)[GBK], unsigned short (*Bs)[GBK], int tid,
    const float4 (&ra)[4], const float4 (&rb)[4]) {
#pragma unroll
  for (int i = 0; i < 4; ++i) {
    int idx = tid + (i << 8);
    int r = idx >> 3;
    int c4 = idx & 7;
    *(uint2*)&As[r][c4 << 2] = pack4(ra[i]);
    *(uint2*)&Bs[r][c4 << 2] = pack4(rb[i]);
  }
}

__global__ __launch_bounds__(256) void gemm_tn_bias(
    const float* __restrict__ A, const float* __restrict__ W,
    const float* __restrict__ bias, float* __restrict__ Y,
    int M, int N, int K) {
  __shared__ __align__(16) unsigned short As[2][GBM][GBK];
  __shared__ __align__(16) unsigned short Bs[2][GBN][GBK];

  const int tid  = threadIdx.x;
  const int lane = tid & 31;
  const int wave = tid >> 5;
  const int wm = wave & 3;   // 4 waves along M (32 rows each)
  const int wn = wave >> 2;  // 2 waves along N (64 cols each)
  const int m0 = blockIdx.x * GBM;
  const int n0 = blockIdx.y * GBN;

  v8f acc[2][4];
#pragma unroll
  for (int i = 0; i < 2; ++i)
#pragma unroll
    for (int j = 0; j < 4; ++j) acc[i][j] = vzero8();

  float4 ra[4], rb[4];
  const int nk = K / GBK;

  // prologue: stage tile 0
  gemm_issue_loads(A, W, M, K, m0, n0, 0, tid, ra, rb);
  gemm_store_tiles(As[0], Bs[0], tid, ra, rb);
  __syncthreads();

  for (int kt = 0; kt < nk; ++kt) {
    const int buf = kt & 1;
    const bool more = (kt + 1) < nk;

    // 1) issue next tile's global loads (in flight under the WMMAs)
    if (more)
      gemm_issue_loads(A, W, M, K, m0, n0, (kt + 1) * GBK, tid, ra, rb);

    // 2) compute on current buffer: 2 (M) x 4 (N) WMMA tiles, K=32
    v16bf af[2], bf4[4];
#pragma unroll
    for (int i = 0; i < 2; ++i)
      af[i] = lds_frag(&As[buf][0][0], 32 * wm + 16 * i, 0, GBK);
#pragma unroll
    for (int j = 0; j < 4; ++j)
      bf4[j] = lds_frag(&Bs[buf][0][0], 64 * wn + 16 * j, 0, GBK);
#pragma unroll
    for (int i = 0; i < 2; ++i)
#pragma unroll
      for (int j = 0; j < 4; ++j)
        acc[i][j] = wmma_bf16(af[i], bf4[j], acc[i][j]);

    // 3) convert + store next tile, then rotate
    if (more)
      gemm_store_tiles(As[buf ^ 1], Bs[buf ^ 1], tid, ra, rb);
    __syncthreads();
  }

  // ---- epilogue: C/D layout -> global, add bias ----
  const int hi = lane >> 4;
  const int ln = lane & 15;
#pragma unroll
  for (int i = 0; i < 2; ++i)
#pragma unroll
    for (int j = 0; j < 4; ++j)
#pragma unroll
      for (int e = 0; e < 8; ++e) {
        int m = m0 + 32 * wm + 16 * i + (hi << 3) + e;
        int n = n0 + 64 * wn + 16 * j + ln;
        if (m < M) Y[(size_t)m * N + n] = acc[i][j][e] + bias[n];
      }
}

// ---------------- fused attention (flash-style, both stages) ----------------
// Q:[Sq,B,H,D]; K/V split as K0(first t0rows rows) + K1(rest), each [.,B,H,D]
// mask:[B,Sq,Ttot] additive; O:[Sq,B,H,D].
// Grid: (B*H, ceil(Sq/64)); 256 threads (8 wave32s); 64-row q-tile, 64-wide T chunks.

__device__ __forceinline__ const float* kv_row(const float* P0, const float* P1,
                                               int t, int t0rows, int b, int h) {
  if (t < t0rows)
    return P0 + ((size_t)(t * B_DIM + b) * H_DIM + h) * D_DIM;
  return P1 + ((size_t)((t - t0rows) * B_DIM + b) * H_DIM + h) * D_DIM;
}

__global__ __launch_bounds__(256) void attn_kernel(
    const float* __restrict__ Q,
    const float* __restrict__ K0, const float* __restrict__ K1,
    const float* __restrict__ V0, const float* __restrict__ V1,
    const float* __restrict__ mask, float* __restrict__ O,
    int Sq, int t0rows, int Ttot) {
  __shared__ __align__(16) unsigned short Qs[64][64];   // bf16 q-tile   [m][d]
  __shared__ __align__(16) unsigned short Ks[64][64];   // bf16 k-tile   [t][d]
  __shared__ __align__(16) unsigned short Vt[64][72];   // bf16 v^T tile [d][t]
  __shared__ __align__(16) float          Sf[64][72];   // mask, then scores [m][t]
  __shared__ __align__(16) unsigned short Ps[64][64];   // bf16 probs    [m][t]
  __shared__ float red[4][64];                          // softmax partials
  __shared__ float mrow[64], lrow[64], srow[64];

  const int tid  = threadIdx.x;
  const int lane = tid & 31;
  const int wave = tid >> 5;
  const int wm = wave & 3;    // 4 waves along M (16 rows each)
  const int wn = wave >> 2;   // 2 waves along N (32 cols each)
  const int bh = blockIdx.x;
  const int b = bh >> 4;      // H = 16
  const int h = bh & 15;
  const int q0 = blockIdx.y * 64;
  const int hi = lane >> 4;
  const int ln = lane & 15;
  // softmax work split: 4 threads per row, 16 cols each
  const int srow_id = tid & 63;
  const int squad   = tid >> 6;

  // ---- load Q tile once (f32 -> bf16) ----
#pragma unroll
  for (int i = 0; i < 4; ++i) {
    int idx = tid + (i << 8);
    int r = idx >> 4;         // 16 float4 per 64-wide row
    int c4 = idx & 15;
    float4 v = {0.f, 0.f, 0.f, 0.f};
    int s = q0 + r;
    if (s < Sq)
      v = *(const float4*)(Q + ((size_t)(s * B_DIM + b) * H_DIM + h) * D_DIM + (c4 << 2));
    *(uint2*)&Qs[r][c4 << 2] = pack4(v);
  }
  if (tid < 64) { mrow[tid] = -1e30f; lrow[tid] = 0.f; }

  v8f o[2];
  o[0] = vzero8();
  o[1] = vzero8();
  __syncthreads();

  for (int t0 = 0; t0 < Ttot; t0 += 64) {
    // ---- async-stage the additive mask tile directly into Sf ----
    // (clamped to stay in-bounds and 16B-aligned; clamped rows/cols are
    //  either overwritten with -1e30 or belong to discarded query rows)
#pragma unroll
    for (int i = 0; i < 4; ++i) {
      int idx = tid + (i << 8);
      int r = idx >> 4;
      int c4 = idx & 15;
      int s = q0 + r;
      if (s >= Sq) s = Sq - 1;
      int t = t0 + (c4 << 2);
      if (t > Ttot - 4) t = Ttot - 4;
      async_load_b128(lds_off32(&Sf[r][c4 << 2]),
                      mask + ((size_t)b * Sq + s) * Ttot + t);
    }
    // ---- load K tile (row-major == B-operand layout for QK^T) ----
#pragma unroll
    for (int i = 0; i < 4; ++i) {
      int idx = tid + (i << 8);
      int r = idx >> 4;
      int c4 = idx & 15;
      int t = t0 + r;
      float4 v = {0.f, 0.f, 0.f, 0.f};
      if (t < Ttot) v = *(const float4*)(kv_row(K0, K1, t, t0rows, b, h) + (c4 << 2));
      *(uint2*)&Ks[r][c4 << 2] = pack4(v);
    }
    // ---- load V transposed: Vt[d][t] (B-operand layout for P*V) ----
#pragma unroll
    for (int i = 0; i < 16; ++i) {
      int idx = tid + (i << 8);
      int r = idx >> 6;       // t within tile
      int dc = idx & 63;      // d
      int t = t0 + r;
      float v = 0.f;
      if (t < Ttot) v = kv_row(V0, V1, t, t0rows, b, h)[dc];
      Vt[dc][r] = f2bf(v);
    }
    wait_async0();
    __syncthreads();

    // ---- S = Q * K^T (per wave: 1x2 tiles, K-dim = D = 64 -> 2 steps) ----
    v8f sc[2];
    sc[0] = vzero8();
    sc[1] = vzero8();
#pragma unroll
    for (int kk = 0; kk < 64; kk += 32) {
      v16bf aq = lds_frag(&Qs[0][0], 16 * wm, kk, 64);
#pragma unroll
      for (int j = 0; j < 2; ++j) {
        v16bf bk = lds_frag(&Ks[0][0], 32 * wn + 16 * j, kk, 64);
        sc[j] = wmma_bf16(aq, bk, sc[j]);
      }
    }
    // scatter scores (C layout), accumulating onto the staged mask in Sf
#pragma unroll
    for (int j = 0; j < 2; ++j)
#pragma unroll
      for (int e = 0; e < 8; ++e) {
        int r = 16 * wm + (hi << 3) + e;
        int c = 32 * wn + 16 * j + ln;
        float val = -1e30f;
        if (t0 + c < Ttot) val = sc[j][e] * SCALING + Sf[r][c];
        Sf[r][c] = val;
      }
    __syncthreads();

    // ---- online softmax: 4 threads per row ----
    {
      float mx = -1e30f;
      const float* row = &Sf[srow_id][squad << 4];
#pragma unroll
      for (int c = 0; c < 16; ++c) mx = fmaxf(mx, row[c]);
      red[squad][srow_id] = mx;
    }
    __syncthreads();
    if (tid < 64) {
      int r = tid;
      float mprev = mrow[r];
      float mx = fmaxf(fmaxf(red[0][r], red[1][r]), fmaxf(red[2][r], red[3][r]));
      mx = fmaxf(mx, mprev);
      srow[r] = __expf(mprev - mx);
      mrow[r] = mx;
    }
    __syncthreads();
    {
      int r = srow_id;
      float mx = mrow[r];
      const float* rowp = &Sf[r][squad << 4];
      unsigned short* pp = &Ps[r][squad << 4];
      float lsum = 0.f;
#pragma unroll
      for (int c = 0; c < 16; ++c) {
        float p = __expf(rowp[c] - mx);
        lsum += p;
        pp[c] = f2bf(p);
      }
      red[squad][r] = lsum;
    }
    __syncthreads();
    if (tid < 64) {
      int r = tid;
      float lsum = red[0][r] + red[1][r] + red[2][r] + red[3][r];
      lrow[r] = lrow[r] * srow[r] + lsum;
    }

    // ---- rescale accumulators, then O += P * V ----
#pragma unroll
    for (int j = 0; j < 2; ++j)
#pragma unroll
      for (int e = 0; e < 8; ++e) {
        int r = 16 * wm + (hi << 3) + e;
        o[j][e] *= srow[r];
      }
#pragma unroll
    for (int kk = 0; kk < 64; kk += 32) {
      v16bf ap = lds_frag(&Ps[0][0], 16 * wm, kk, 64);
#pragma unroll
      for (int j = 0; j < 2; ++j) {
        v16bf bv = lds_frag(&Vt[0][0], 32 * wn + 16 * j, kk, 72);
        o[j] = wmma_bf16(ap, bv, o[j]);
      }
    }
    __syncthreads();
  }

  // ---- epilogue: O / l -> [Sq,B,H,D] ----
#pragma unroll
  for (int j = 0; j < 2; ++j)
#pragma unroll
    for (int e = 0; e < 8; ++e) {
      int r = 16 * wm + (hi << 3) + e;
      int s = q0 + r;
      int dc = 32 * wn + 16 * j + ln;
      if (s < Sq)
        O[((size_t)(s * B_DIM + b) * H_DIM + h) * D_DIM + dc] = o[j][e] / lrow[r];
    }
}

// ---------------- host-side orchestration ----------------
extern "C" void kernel_launch(void* const* d_in, const int* in_sizes, int n_in,
                              void* d_out, int out_size, void* d_ws, size_t ws_size,
                              hipStream_t stream) {
  (void)in_sizes; (void)n_in; (void)out_size; (void)ws_size;

  const float* sum_x   = (const float*)d_in[0];
  const float* reg_x   = (const float*)d_in[1];
  // d_in[2] sum_token_ids (unused by reference), d_in[3]/[4] lengths (compile-time here)
  const float* mask_sx = (const float*)d_in[5];
  const float* mask_rx = (const float*)d_in[6];
  const float* Wrq = (const float*)d_in[7];  const float* brq = (const float*)d_in[8];
  const float* Wrk = (const float*)d_in[9];  const float* brk = (const float*)d_in[10];
  const float* Wrv = (const float*)d_in[11]; const float* brv = (const float*)d_in[12];
  const float* Wro = (const float*)d_in[13]; const float* bro = (const float*)d_in[14];
  const float* Wsq = (const float*)d_in[15]; const float* bsq = (const float*)d_in[16];
  const float* Wsk = (const float*)d_in[17]; const float* bsk = (const float*)d_in[18];
  const float* Wsv = (const float*)d_in[19]; const float* bsv = (const float*)d_in[20];
  const float* Wso = (const float*)d_in[21]; const float* bso = (const float*)d_in[22];
  const float* Wk2 = (const float*)d_in[23]; const float* bk2 = (const float*)d_in[24];
  const float* Wv2 = (const float*)d_in[25]; const float* bv2 = (const float*)d_in[26];

  const int MR = R_REG * B_DIM;   // 4096
  const int MS = S_SUM * B_DIM;   // 64

  // fp32 workspace layout
  float* ws = (float*)d_ws;
  const size_t SZR = (size_t)MR * E_DIM;   // 4M floats
  const size_t SZS = (size_t)MS * E_DIM;   // 64K floats
  float* reg_k  = ws;
  float* reg_v  = reg_k  + SZR;
  float* reg_q  = reg_v  + SZR;
  float* reg_at = reg_q  + SZR;
  float* sum_q  = reg_at + SZR;
  float* sum_k  = sum_q  + SZS;
  float* sum_v  = sum_k  + SZS;
  float* sum_x2 = sum_v  + SZS;
  float* sum_k2 = sum_x2 + SZS;
  float* sum_v2 = sum_k2 + SZS;

  float* sum_out = (float*)d_out;
  float* reg_out = sum_out + SZS;

  dim3 blk(256);
  dim3 gR((MR + GBM - 1) / GBM, E_DIM / GBN);  // (32, 8)
  dim3 gS((MS + GBM - 1) / GBM, E_DIM / GBN);  // (1, 8)
  dim3 a1(B_DIM * H_DIM, (S_SUM + 63) / 64);   // (32, 1)
  dim3 a2(B_DIM * H_DIM, (R_REG + 63) / 64);   // (32, 32)

  // regular-stream K/V projections
  gemm_tn_bias<<<gR, blk, 0, stream>>>(reg_x, Wrk, brk, reg_k, MR, E_DIM, E_DIM);
  gemm_tn_bias<<<gR, blk, 0, stream>>>(reg_x, Wrv, brv, reg_v, MR, E_DIM, E_DIM);
  // summary-stream Q/K/V projections
  gemm_tn_bias<<<gS, blk, 0, stream>>>(sum_x, Wsq, bsq, sum_q, MS, E_DIM, E_DIM);
  gemm_tn_bias<<<gS, blk, 0, stream>>>(sum_x, Wsk, bsk, sum_k, MS, E_DIM, E_DIM);
  gemm_tn_bias<<<gS, blk, 0, stream>>>(sum_x, Wsv, bsv, sum_v, MS, E_DIM, E_DIM);
  // stage 1: summary queries over [sum_k ; reg_k]
  attn_kernel<<<a1, blk, 0, stream>>>(sum_q, sum_k, reg_k, sum_v, reg_v,
                                      mask_sx, sum_x2, S_SUM, S_SUM, T_TOT);
  // second-level projections from aggregated summaries
  gemm_tn_bias<<<gS, blk, 0, stream>>>(sum_x2, Wk2, bk2, sum_k2, MS, E_DIM, E_DIM);
  gemm_tn_bias<<<gS, blk, 0, stream>>>(sum_x2, Wv2, bv2, sum_v2, MS, E_DIM, E_DIM);
  // regular queries
  gemm_tn_bias<<<gR, blk, 0, stream>>>(reg_x, Wrq, brq, reg_q, MR, E_DIM, E_DIM);
  // stage 2: regular queries over [sum_k2 ; reg_k]
  attn_kernel<<<a2, blk, 0, stream>>>(reg_q, sum_k2, reg_k, sum_v2, reg_v,
                                      mask_rx, reg_at, R_REG, S_SUM, T_TOT);
  // output projections
  gemm_tn_bias<<<gS, blk, 0, stream>>>(sum_x2, Wso, bso, sum_out, MS, E_DIM, E_DIM);
  gemm_tn_bias<<<gR, blk, 0, stream>>>(reg_at, Wro, bro, reg_out, MR, E_DIM, E_DIM);
}
